// MSDeformAttnPixelDecoder_77464030150786
// MI455X (gfx1250) — matrix operand order repaired and checked by reading, not compiled, not run on hardware
//
#include <hip/hip_runtime.h>
#include <hip/hip_bf16.h>

// MSDeformAttnPixelDecoder for gfx1250 (MI455X).
// GEMMs via v_wmma_f32_16x16x32_f16 (wave32, 16x16x32), fp32 elsewhere.
// ffn2 A-tile staging uses global_load_async_to_lds_b64 (ASYNCcnt path).

typedef __attribute__((ext_vector_type(16))) _Float16 v16h;
typedef __attribute__((ext_vector_type(8)))  _Float16 v8h;
typedef __attribute__((ext_vector_type(4)))  _Float16 v4h;
typedef __attribute__((ext_vector_type(8)))  float    v8f;
typedef __attribute__((ext_vector_type(4)))  float    v4f;
typedef _Float16 half_t;

constexpr int EP_STORE    = 0;   // C = acc (+bias), strided store (row-major or NCHW)
constexpr int EP_RELU_F16 = 1;   // Ch = (half) relu(acc + bias)
constexpr int EP_RESID_LN = 2;   // C = LayerNorm(acc + bias + resid), N must be 256

constexpr int A_F32_ROW   = 0;   // A fp32, row-major (a_rs = row stride, a_cs = 1)
constexpr int A_F32_STRID = 1;   // A fp32, column-strided (NCHW feature: a_rs=1, a_cs=HW)
constexpr int A_F16_ROW   = 2;   // A fp16, row-major (staged via async global->LDS)

// LDS A tile: 32 rows x 32 halves, 48-half (96 B) pitch -> every 16B chunk aligned.
#define A_PITCH 48

// A fragment (M=16 x K=32): lane m = lane&15; lanes 0-15: K in {0..15},
// lanes 16-31: K in {16..31}.  Two aligned 16B LDS loads (ds_load_b128).
__device__ inline v16h load_a_frag(const half_t* As, int m_local, int lane) {
    int o = (lane & 16) ? 16 : 0;
    const half_t* row = As + (size_t)m_local * A_PITCH;
    union { v16h v; v8h h[2]; } u;
    u.h[0] = *(const v8h*)(row + o);
    u.h[1] = *(const v8h*)(row + o + 8);
    return u.v;
}

// ---------------- Generic WMMA GEMM ----------------------------------------
// Block: 256 thr = 8 waves. Output tile 32 rows x 256 cols (wave: 32x32 = 2x2 frags).
// REQUIRES: M % 32 == 0, K % 32 == 0 (true for every launch in this pipeline).
// Wt: fp16 N-major [N][K]. grid = (M/32, ceil(N/256), batches).
template<int EP, int AMODE>
__global__ __launch_bounds__(256) void gemm_wmma(
    const float* Af, const half_t* Ah, long a_rs, long a_cs, long a_bs,
    const half_t* Wt, const float* bias,
    float* C, half_t* Ch, long c_rs, long c_cs, long c_bs,
    int M, int N, int K,
    const float* resid, const float* ln_s, const float* ln_b)
{
    __shared__ half_t As[32][A_PITCH];
    if constexpr (AMODE == A_F16_ROW) Ah += (size_t)blockIdx.z * a_bs;
    else                              Af += (size_t)blockIdx.z * a_bs;
    if (C) C += (size_t)blockIdx.z * c_bs;

    const int row0 = blockIdx.x * 32;
    const int lane = threadIdx.x & 31;
    const int wave = threadIdx.x >> 5;
    const int n_base = blockIdx.y * 256 + wave * 32;

    // per-thread staging pointers (advanced additively each K step)
    const float*  apf = nullptr;
    const half_t* aph = nullptr;
    int r_st, kq_st;
    if constexpr (AMODE == A_F32_ROW) {
        r_st = threadIdx.x >> 3; kq_st = (threadIdx.x & 7) * 4;
        apf = Af + (size_t)(row0 + r_st) * a_rs + kq_st;
    } else if constexpr (AMODE == A_F16_ROW) {
        r_st = threadIdx.x >> 3; kq_st = (threadIdx.x & 7) * 4;
        aph = Ah + (size_t)(row0 + r_st) * a_rs + kq_st;
    } else {
        r_st = threadIdx.x & 31; kq_st = threadIdx.x >> 5;
        apf = Af + (size_t)(row0 + r_st) * a_rs + (size_t)kq_st * a_cs;
    }
    // per-wave B fragment pointers (N-major weights, 64B-aligned rows)
    const half_t* wp0 = Wt + (size_t)(n_base + (lane & 15)) * K + ((lane & 16) ? 16 : 0);
    const half_t* wp1 = wp0 + (size_t)16 * K;

    v8f acc[2][2] = {};

    for (int kc = 0; kc < K; kc += 32) {
        // ---- stage A tile (32 rows x 32 k) into LDS as fp16, branch-free
        if constexpr (AMODE == A_F32_ROW) {
            v4f x = *(const v4f*)apf;
            __builtin_prefetch(apf + 32, 0, 0);   // next K tile -> global_prefetch_b8
            v4h hx;
            hx[0] = (half_t)x[0]; hx[1] = (half_t)x[1];
            hx[2] = (half_t)x[2]; hx[3] = (half_t)x[3];
            *(v4h*)&As[r_st][kq_st] = hx;
            apf += 32;
        } else if constexpr (AMODE == A_F16_ROW) {
            // CDNA5 async copy: 8 bytes/lane global -> LDS, tracked by ASYNCcnt.
            // Generic LDS pointer low 32 bits == LDS byte offset (ISA 10.2).
            unsigned lds_a = (unsigned)(size_t)&As[r_st][kq_st];
            unsigned long long ga = (unsigned long long)(size_t)aph;
            asm volatile("global_load_async_to_lds_b64 %0, %1, off"
                         :: "v"(lds_a), "v"(ga) : "memory");
            asm volatile("s_wait_asynccnt 0" ::: "memory");
            aph += 32;
        } else { // A_F32_STRID: lane-fast over rows; additive channel offsets
#pragma unroll
            for (int j = 0; j < 4; ++j)
                As[r_st][kq_st + j * 8] = (half_t)apf[(size_t)(j * 8) * a_cs];
            apf += (size_t)32 * a_cs;
        }
        __syncthreads();

        if (n_base < N) { // wave-uniform branch: EXEC stays all-ones for WMMA
            v16h a0 = load_a_frag(&As[0][0], (lane & 15), lane);
            v16h a1 = load_a_frag(&As[0][0], 16 + (lane & 15), lane);
            v16h b0 = *(const v16h*)wp0;
            v16h b1 = *(const v16h*)wp1;
            acc[0][0] = __builtin_amdgcn_wmma_f32_16x16x32_f16(false, a0, false, b0, (short)0, acc[0][0], false, false);
            acc[0][1] = __builtin_amdgcn_wmma_f32_16x16x32_f16(false, a0, false, b1, (short)0, acc[0][1], false, false);
            acc[1][0] = __builtin_amdgcn_wmma_f32_16x16x32_f16(false, a1, false, b0, (short)0, acc[1][0], false, false);
            acc[1][1] = __builtin_amdgcn_wmma_f32_16x16x32_f16(false, a1, false, b1, (short)0, acc[1][1], false, false);
        }
        wp0 += 32; wp1 += 32;
        __syncthreads();
    }

    // ---- epilogues.  D layout: VGPR r, lane l -> M = r + (l>=16 ? 8:0), N = l&15.
    if constexpr (EP == EP_STORE) {
        if (n_base < N) {
#pragma unroll
            for (int mt = 0; mt < 2; ++mt)
#pragma unroll
            for (int nt = 0; nt < 2; ++nt) {
                int n = n_base + nt * 16 + (lane & 15);
                float bv = bias ? bias[n] : 0.f;
#pragma unroll
                for (int r = 0; r < 8; ++r) {
                    int m = row0 + mt * 16 + ((lane & 16) ? 8 : 0) + r;
                    C[(size_t)m * c_rs + (size_t)n * c_cs] = acc[mt][nt][r] + bv;
                }
            }
        }
    } else if constexpr (EP == EP_RELU_F16) {
        if (n_base < N) {
#pragma unroll
            for (int mt = 0; mt < 2; ++mt)
#pragma unroll
            for (int nt = 0; nt < 2; ++nt) {
                int n = n_base + nt * 16 + (lane & 15);
                float bv = bias[n];
#pragma unroll
                for (int r = 0; r < 8; ++r) {
                    int m = row0 + mt * 16 + ((lane & 16) ? 8 : 0) + r;
                    float v = fmaxf(acc[mt][nt][r] + bv, 0.f);
                    Ch[(size_t)m * N + n] = (half_t)v;
                }
            }
        }
    } else { // EP_RESID_LN, requires N==256, gridDim.y==1
        __shared__ float Bb[32][257];
        __shared__ float red_s[32][8], red_q[32][8];
        __shared__ float mu_s[32], rstd_s[32];
#pragma unroll
        for (int mt = 0; mt < 2; ++mt)
#pragma unroll
        for (int nt = 0; nt < 2; ++nt) {
            int n = n_base + nt * 16 + (lane & 15);
            float bv = bias[n];
#pragma unroll
            for (int r = 0; r < 8; ++r) {
                int ml = mt * 16 + ((lane & 16) ? 8 : 0) + r;
                int m = row0 + ml;
                Bb[ml][n] = acc[mt][nt][r] + bv + resid[(size_t)m * 256 + n];
            }
        }
        __syncthreads();
        int rrow = threadIdx.x >> 3, sub = threadIdx.x & 7;
        float s = 0.f, q = 0.f;
#pragma unroll 8
        for (int j = 0; j < 32; ++j) { float v = Bb[rrow][sub * 32 + j]; s += v; q += v * v; }
        red_s[rrow][sub] = s; red_q[rrow][sub] = q;
        __syncthreads();
        if (sub == 0) {
            float S = 0.f, Q = 0.f;
#pragma unroll
            for (int i = 0; i < 8; ++i) { S += red_s[rrow][i]; Q += red_q[rrow][i]; }
            float mu = S * (1.f / 256.f);
            float var = Q * (1.f / 256.f) - mu * mu;
            mu_s[rrow] = mu; rstd_s[rrow] = rsqrtf(var + 1e-5f);
        }
        __syncthreads();
        int m = row0 + rrow;
        float mu = mu_s[rrow], rs = rstd_s[rrow];
#pragma unroll 8
        for (int j = 0; j < 32; ++j) {
            int n = sub * 32 + j;
            C[(size_t)m * 256 + n] = (Bb[rrow][n] - mu) * rs * ln_s[n] + ln_b[n];
        }
    }
}

// ---------------- 3x3 conv as 9 tap-shifted WMMA GEMMs ----------------------
// X,C: per-batch NCHW (grid.z = batch). Wt9: fp16, 9 taps of [256][256] N-major.
__global__ __launch_bounds__(256) void conv3x3_wmma(
    const float* X, const half_t* Wt9, float* C, int H, int W)
{
    __shared__ half_t As[32][A_PITCH];
    const int HW = H * W;
    X += (size_t)blockIdx.z * 256 * HW;
    C += (size_t)blockIdx.z * 256 * HW;
    const int row0 = blockIdx.x * 32;
    const int lane = threadIdx.x & 31;
    const int wave = threadIdx.x >> 5;
    const int n_base = wave * 32;

    const int r  = threadIdx.x & 31;   // staging row (spatial position)
    const int c0 = threadIdx.x >> 5;   // staging column group (channel)
    const int m_st = row0 + r;
    const int ys = m_st / W, xs = m_st % W;

    v8f acc[2][2] = {};
    for (int tap = 0; tap < 9; ++tap) {
        int dy = tap / 3 - 1, dx = tap % 3 - 1;
        int yy = ys + dy, xx = xs + dx;
        int yc = min(max(yy, 0), H - 1), xc = min(max(xx, 0), W - 1);
        float okf = (yy == yc && xx == xc) ? 1.f : 0.f;  // branchless zero-pad
        const float* xp = X + (size_t)c0 * HW + (size_t)yc * W + xc;
        const half_t* wp0 = Wt9 + (size_t)tap * 256 * 256
                          + (size_t)(n_base + (lane & 15)) * 256 + ((lane & 16) ? 16 : 0);
        const half_t* wp1 = wp0 + (size_t)16 * 256;
        for (int kc = 0; kc < 256; kc += 32) {
#pragma unroll
            for (int j = 0; j < 4; ++j)
                As[r][c0 + j * 8] = (half_t)(xp[(size_t)(j * 8) * HW] * okf);
            xp += (size_t)32 * HW;
            __syncthreads();
            v16h a0 = load_a_frag(&As[0][0], (lane & 15), lane);
            v16h a1 = load_a_frag(&As[0][0], 16 + (lane & 15), lane);
            v16h b0 = *(const v16h*)wp0;
            v16h b1 = *(const v16h*)wp1;
            acc[0][0] = __builtin_amdgcn_wmma_f32_16x16x32_f16(false, a0, false, b0, (short)0, acc[0][0], false, false);
            acc[0][1] = __builtin_amdgcn_wmma_f32_16x16x32_f16(false, a0, false, b1, (short)0, acc[0][1], false, false);
            acc[1][0] = __builtin_amdgcn_wmma_f32_16x16x32_f16(false, a1, false, b0, (short)0, acc[1][0], false, false);
            acc[1][1] = __builtin_amdgcn_wmma_f32_16x16x32_f16(false, a1, false, b1, (short)0, acc[1][1], false, false);
            wp0 += 32; wp1 += 32;
            __syncthreads();
        }
    }
#pragma unroll
    for (int mt = 0; mt < 2; ++mt)
#pragma unroll
    for (int nt = 0; nt < 2; ++nt) {
        int n = n_base + nt * 16 + (lane & 15);
#pragma unroll
        for (int rr = 0; rr < 8; ++rr) {
            int m = row0 + mt * 16 + ((lane & 16) ? 8 : 0) + rr; // HW % 32 == 0
            C[(size_t)n * HW + m] = acc[mt][nt][rr];
        }
    }
}

// ---------------- weight conversion fp32 -> fp16 (with layout remap) --------
__global__ void cvt_w_kernel(const float* src, half_t* dst, int N, int K,
                             long sn, long sk, long off) {
    int i = blockIdx.x * 256 + threadIdx.x;
    if (i >= N * K) return;
    int n = i / K, k = i % K;
    dst[i] = (half_t)src[(long)n * sn + (long)k * sk + off];
}

// ---------------- GroupNorm (32 groups over 256 ch), optional ReLU ----------
__global__ __launch_bounds__(256) void groupnorm_k(
    const float* src, float* dst, const float* gamma, const float* beta,
    int HW, int relu)
{
    __shared__ float ss[256], sq[256];
    int b = blockIdx.x >> 5, g = blockIdx.x & 31;
    const float* p = src + ((size_t)b * 256 + g * 8) * HW;
    float* d = dst + ((size_t)b * 256 + g * 8) * HW;
    int n = 8 * HW;
    float s = 0.f, q = 0.f;
    for (int i = threadIdx.x; i < n; i += 256) { float v = p[i]; s += v; q += v * v; }
    ss[threadIdx.x] = s; sq[threadIdx.x] = q;
    __syncthreads();
    for (int st = 128; st > 0; st >>= 1) {
        if (threadIdx.x < st) { ss[threadIdx.x] += ss[threadIdx.x + st]; sq[threadIdx.x] += sq[threadIdx.x + st]; }
        __syncthreads();
    }
    float mu = ss[0] / n;
    float rstd = rsqrtf(sq[0] / n - mu * mu + 1e-5f);
    for (int i = threadIdx.x; i < n; i += 256) {
        int c = g * 8 + i / HW;
        float v = (p[i] - mu) * rstd * gamma[c] + beta[c];
        if (relu) v = fmaxf(v, 0.f);
        d[i] = v;
    }
}

// ---------------- bilinear resize(prev) + add into lat (in place) -----------
__global__ void upsample_add_k(const float* prev, float* lat,
                               int Hp, int Wp, int H, int W) {
    int total = 2 * 256 * H * W;
    int i = blockIdx.x * 256 + threadIdx.x;
    if (i >= total) return;
    int hw = i % (H * W);
    int y = hw / W, x = hw % W;
    int bc = i / (H * W);
    float sy = (y + 0.5f) * (float)Hp / (float)H - 0.5f;
    float sx = (x + 0.5f) * (float)Wp / (float)W - 0.5f;
    float fy = sy - floorf(sy), fx = sx - floorf(sx);
    int y0 = (int)floorf(sy), x0 = (int)floorf(sx);
    int y1 = min(y0 + 1, Hp - 1), x1 = min(x0 + 1, Wp - 1);
    y0 = max(y0, 0); x0 = max(x0, 0);
    const float* pb = prev + (size_t)bc * Hp * Wp;
    float v = (1.f - fy) * ((1.f - fx) * pb[y0 * Wp + x0] + fx * pb[y0 * Wp + x1])
            +        fy  * ((1.f - fx) * pb[y1 * Wp + x0] + fx * pb[y1 * Wp + x1]);
    lat[i] += v;
}

// ---------------- sine positional encoding + reference points ---------------
__global__ void posref_k(float* pos, float* ref) {
    int i = blockIdx.x * 256 + threadIdx.x;
    if (i >= 5440 * 256) return;
    int s = i >> 8, c = i & 255;
    int base, h, w;
    if      (s < 4096) { base = 0;    h = 64; w = 64; }
    else if (s < 5120) { base = 4096; h = 32; w = 32; }
    else if (s < 5376) { base = 5120; h = 16; w = 16; }
    else               { base = 5376; h = 8;  w = 8;  }
    int r = s - base;
    int y = r / w, x = r % w;
    const float scale = 2.0f * 3.14159f;
    float fy = (y + 1.0f) / ((float)h + 1e-6f) * scale;
    float fx = (x + 1.0f) / ((float)w + 1e-6f) * scale;
    float bv = (c < 128) ? fy : fx;
    int cc = c & 127, m = cc >> 1;
    float t = powf(10000.f, (2.f * m) * (1.f / 128.f));
    float a = bv / t;
    pos[i] = (cc & 1) ? cosf(a) : sinf(a);
    if (c == 0) {
        ref[s * 2 + 0] = (x + 0.5f) / (float)w;
        ref[s * 2 + 1] = (y + 0.5f) / (float)h;
    }
}

// ---------------- q = src + pos (broadcast over batch) ----------------------
__global__ void addpos_k(const float* src, const float* pos, float* q, int total, int perb) {
    int i = blockIdx.x * 256 + threadIdx.x;
    if (i >= total) return;
    q[i] = src[i] + pos[i % perb];
}

// ---------------- softmax over 16 (per b,s,head) ----------------------------
__global__ void softmax16_k(const float* in, float* out) {
    int i = blockIdx.x * 256 + threadIdx.x;
    if (i >= 2 * 5440 * 8) return;
    const float* p = in + (size_t)i * 16;
    float mx = p[0];
    for (int j = 1; j < 16; ++j) mx = fmaxf(mx, p[j]);
    float e[16], s = 0.f;
    for (int j = 0; j < 16; ++j) { e[j] = expf(p[j] - mx); s += e[j]; }
    float inv = 1.f / s;
    float* o = out + (size_t)i * 16;
    for (int j = 0; j < 16; ++j) o[j] = e[j] * inv;
}

// ---------------- MS deformable sampling: wave per (b,s,head), lane = d -----
__global__ __launch_bounds__(256) void msdeform_sampler(
    const float* value, const float* off, const float* aw,
    const float* ref, float* out)
{
    const int S = 5440;
    int wave = threadIdx.x >> 5, lane = threadIdx.x & 31;
    int gid = blockIdx.x * 8 + wave;
    if (gid >= 2 * S * 8) return;
    int head = gid & 7;
    int s = (gid >> 3) % S;
    int b = gid / (8 * S);
    const int lvl_start[4] = {0, 4096, 5120, 5376};
    const int lvl_h[4] = {64, 32, 16, 8};
    const int lvl_w[4] = {64, 32, 16, 8};
    float rx = ref[s * 2 + 0], ry = ref[s * 2 + 1];
    size_t rowbase = (size_t)b * S + s;
    const float* offp = off + rowbase * 256 + head * 32;
    const float* awp  = aw  + rowbase * 128 + head * 16;
    float acc = 0.f;
#pragma unroll
    for (int l = 0; l < 4; ++l) {
        int h = lvl_h[l], w = lvl_w[l];
        const float* vbase = value + ((size_t)b * S + lvl_start[l]) * 256 + head * 32 + lane;
#pragma unroll
        for (int p = 0; p < 4; ++p) {
            float ox = offp[l * 8 + p * 2 + 0];
            float oy = offp[l * 8 + p * 2 + 1];
            float x = (rx + ox / (float)w) * w - 0.5f;
            float y = (ry + oy / (float)h) * h - 0.5f;
            float x0 = floorf(x), y0 = floorf(y);
            int x0i = (int)x0, y0i = (int)y0;
            float wx1 = x - x0, wy1 = y - y0;
            float wx0 = 1.f - wx1, wy0 = 1.f - wy1;
            float a = awp[l * 4 + p];
            float sv = 0.f;
#pragma unroll
            for (int cy = 0; cy < 2; ++cy)
#pragma unroll
            for (int cx = 0; cx < 2; ++cx) {
                int xi = x0i + cx, yi = y0i + cy;
                if (xi >= 0 && xi < w && yi >= 0 && yi < h) {
                    float wgt = (cx ? wx1 : wx0) * (cy ? wy1 : wy0);
                    sv += wgt * vbase[(size_t)(yi * w + xi) * 256];
                }
            }
            acc += a * sv;
        }
    }
    out[rowbase * 256 + head * 32 + lane] = acc;
}

// ---------------- layout shuffles -------------------------------------------
__global__ void chw_to_seq_k(const float* x, float* src, int HW, int start) {
    int total = 2 * 256 * HW;
    int i = blockIdx.x * 256 + threadIdx.x;
    if (i >= total) return;
    int b = i / (256 * HW);
    int r = i % (256 * HW);
    int c = r / HW, hw = r % HW;
    src[((size_t)b * 5440 + start + hw) * 256 + c] = x[i];
}
__global__ void seq_to_chw_k(const float* src, float* x, int HW, int start) {
    int total = 2 * 256 * HW;
    int i = blockIdx.x * 256 + threadIdx.x;
    if (i >= total) return;
    int b = i / (256 * HW);
    int r = i % (256 * HW);
    int c = r / HW, hw = r % HW;
    x[i] = src[((size_t)b * 5440 + start + hw) * 256 + c];
}

// ---------------- host driver ------------------------------------------------
extern "C" void kernel_launch(void* const* d_in, const int* in_sizes, int n_in,
                              void* d_out, int out_size, void* d_ws, size_t ws_size,
                              hipStream_t stream) {
    (void)in_sizes; (void)n_in; (void)out_size; (void)ws_size;
    const float* feat[4]  = {(const float*)d_in[0], (const float*)d_in[1],
                             (const float*)d_in[2], (const float*)d_in[3]};
    const float* lat_w[4] = {(const float*)d_in[4], (const float*)d_in[5],
                             (const float*)d_in[6], (const float*)d_in[7]};
    const float* lat_gn_s = (const float*)d_in[8];
    const float* lat_gn_b = (const float*)d_in[9];
    const float* out_w    = (const float*)d_in[10];
    const float* out_gn_s = (const float*)d_in[11];
    const float* out_gn_b = (const float*)d_in[12];
    const float* off_w  = (const float*)d_in[13]; const float* off_b  = (const float*)d_in[14];
    const float* aw_w   = (const float*)d_in[15]; const float* aw_b   = (const float*)d_in[16];
    const float* val_w  = (const float*)d_in[17]; const float* val_b  = (const float*)d_in[18];
    const float* proj_w = (const float*)d_in[19]; const float* proj_b = (const float*)d_in[20];
    const float* ffn1_w = (const float*)d_in[21]; const float* ffn1_b = (const float*)d_in[22];
    const float* ffn2_w = (const float*)d_in[23]; const float* ffn2_b = (const float*)d_in[24];
    const float* ln1_s = (const float*)d_in[25]; const float* ln1_b = (const float*)d_in[26];
    const float* ln2_s = (const float*)d_in[27]; const float* ln2_b = (const float*)d_in[28];
    const float* mask_w = (const float*)d_in[29]; const float* mask_b = (const float*)d_in[30];

    const int S = 5440, M = 10880;
    const int hs[4] = {64, 32, 16, 8}, wd[4] = {64, 32, 16, 8};
    const int HW[4] = {4096, 1024, 256, 64};
    const int starts[4] = {0, 4096, 5120, 5376};
    const int cins[4] = {256, 512, 1024, 2048};

    size_t woff = 0;
    auto alloc = [&](size_t bytes) -> char* {
        char* p = (char*)d_ws + woff;
        woff += (bytes + 255) & ~(size_t)255;
        return p;
    };
    // fp16 weights
    half_t* h_lat[4];
    for (int l = 0; l < 4; ++l) h_lat[l] = (half_t*)alloc((size_t)256 * cins[l] * 2);
    half_t* h_outw = (half_t*)alloc((size_t)4 * 9 * 256 * 256 * 2);
    half_t* h_val  = (half_t*)alloc(256 * 256 * 2);
    half_t* h_off  = (half_t*)alloc(256 * 256 * 2);
    half_t* h_aw   = (half_t*)alloc(128 * 256 * 2);
    half_t* h_proj = (half_t*)alloc(256 * 256 * 2);
    half_t* h_ffn1 = (half_t*)alloc((size_t)2048 * 256 * 2);
    half_t* h_ffn2 = (half_t*)alloc((size_t)256 * 2048 * 2);
    half_t* h_mask = (half_t*)alloc(256 * 256 * 2);
    // fp32 activations
    float* f_lat  = (float*)alloc((size_t)2 * 256 * 4096 * 4);
    float* f_conv = (float*)alloc((size_t)2 * 256 * 4096 * 4);
    float* f_ms[4];
    for (int l = 0; l < 4; ++l) f_ms[l] = (float*)alloc((size_t)2 * 256 * HW[l] * 4);
    float* f_srcA = (float*)alloc((size_t)M * 256 * 4);
    float* f_srcB = (float*)alloc((size_t)M * 256 * 4);
    float* f_mid  = (float*)alloc((size_t)M * 256 * 4);
    float* f_pos  = (float*)alloc((size_t)S * 256 * 4);
    float* f_ref  = (float*)alloc((size_t)S * 2 * 4);
    float* f_q    = (float*)alloc((size_t)M * 256 * 4);
    float* f_val  = (float*)alloc((size_t)M * 256 * 4);
    float* f_off  = (float*)alloc((size_t)M * 256 * 4);
    float* f_awl  = (float*)alloc((size_t)M * 128 * 4);
    float* f_aw   = (float*)alloc((size_t)M * 128 * 4);
    float* f_attn = (float*)alloc((size_t)M * 256 * 4);
    half_t* h_act = (half_t*)alloc((size_t)M * 2048 * 2);

    auto cvt = [&](const float* src, half_t* dst, int N, int K, long sn, long sk, long off) {
        cvt_w_kernel<<<(N * K + 255) / 256, 256, 0, stream>>>(src, dst, N, K, sn, sk, off);
    };
    // weight prep: linear weights (K,N)->[N][K]; conv weights (O,C)->[N=O][K=C]
    for (int l = 0; l < 4; ++l) cvt(lat_w[l], h_lat[l], 256, cins[l], cins[l], 1, 0);
    for (int l = 0; l < 4; ++l)
        for (int tap = 0; tap < 9; ++tap)
            cvt(out_w, h_outw + ((size_t)l * 9 + tap) * 256 * 256,
                256, 256, 2304, 9, (long)l * 589824 + tap);
    cvt(val_w,  h_val,  256, 256, 1, 256, 0);
    cvt(off_w,  h_off,  256, 256, 1, 256, 0);
    cvt(aw_w,   h_aw,   128, 256, 1, 128, 0);
    cvt(proj_w, h_proj, 256, 256, 1, 256, 0);
    cvt(ffn1_w, h_ffn1, 2048, 256, 1, 2048, 0);
    cvt(ffn2_w, h_ffn2, 256, 2048, 1, 256, 0);
    cvt(mask_w, h_mask, 256, 256, 256, 1, 0);

    posref_k<<<5440, 256, 0, stream>>>(f_pos, f_ref);

    // ---- FPN neck
    for (int l = 0; l < 4; ++l) {
        int hw = HW[l];
        // lateral 1x1 conv: out_t[hw,o] = X_t[hw,c] * W[o,c], NCHW in & out
        gemm_wmma<EP_STORE, A_F32_STRID><<<dim3(hw / 32, 1, 2), 256, 0, stream>>>(
            feat[l], nullptr, 1, hw, (long)cins[l] * hw,
            h_lat[l], nullptr,
            f_lat, nullptr, 1, hw, 256L * hw,
            hw, 256, cins[l], nullptr, nullptr, nullptr);
        groupnorm_k<<<64, 256, 0, stream>>>(f_lat, f_lat,
            lat_gn_s + l * 256, lat_gn_b + l * 256, hw, 0);
        if (l > 0)
            upsample_add_k<<<(2 * 256 * hw + 255) / 256, 256, 0, stream>>>(
                f_ms[l - 1], f_lat, hs[l - 1], wd[l - 1], hs[l], wd[l]);
        conv3x3_wmma<<<dim3(hw / 32, 1, 2), 256, 0, stream>>>(
            f_lat, h_outw + (size_t)l * 9 * 256 * 256, f_conv, hs[l], wd[l]);
        groupnorm_k<<<64, 256, 0, stream>>>(f_conv, f_ms[l],
            out_gn_s + l * 256, out_gn_b + l * 256, hw, 1);
        chw_to_seq_k<<<(2 * 256 * hw + 255) / 256, 256, 0, stream>>>(
            f_ms[l], f_srcA, hw, starts[l]);
    }

    // ---- 6 encoder layers, shared weights
    float* cur = f_srcA;
    float* nxt = f_srcB;
    for (int layer = 0; layer < 6; ++layer) {
        addpos_k<<<(M * 256 + 255) / 256, 256, 0, stream>>>(cur, f_pos, f_q, M * 256, S * 256);
        gemm_wmma<EP_STORE, A_F32_ROW><<<dim3(M / 32, 1, 1), 256, 0, stream>>>(
            cur, nullptr, 256, 1, 0, h_val, val_b,
            f_val, nullptr, 256, 1, 0, M, 256, 256, nullptr, nullptr, nullptr);
        gemm_wmma<EP_STORE, A_F32_ROW><<<dim3(M / 32, 1, 1), 256, 0, stream>>>(
            f_q, nullptr, 256, 1, 0, h_off, off_b,
            f_off, nullptr, 256, 1, 0, M, 256, 256, nullptr, nullptr, nullptr);
        gemm_wmma<EP_STORE, A_F32_ROW><<<dim3(M / 32, 1, 1), 256, 0, stream>>>(
            f_q, nullptr, 256, 1, 0, h_aw, aw_b,
            f_awl, nullptr, 128, 1, 0, M, 128, 256, nullptr, nullptr, nullptr);
        softmax16_k<<<(2 * S * 8 + 255) / 256, 256, 0, stream>>>(f_awl, f_aw);
        msdeform_sampler<<<M, 256, 0, stream>>>(f_val, f_off, f_aw, f_ref, f_attn);
        gemm_wmma<EP_RESID_LN, A_F32_ROW><<<dim3(M / 32, 1, 1), 256, 0, stream>>>(
            f_attn, nullptr, 256, 1, 0, h_proj, proj_b,
            f_mid, nullptr, 256, 1, 0, M, 256, 256, cur, ln1_s, ln1_b);
        gemm_wmma<EP_RELU_F16, A_F32_ROW><<<dim3(M / 32, 8, 1), 256, 0, stream>>>(
            f_mid, nullptr, 256, 1, 0, h_ffn1, ffn1_b,
            nullptr, h_act, 2048, 1, 0, M, 2048, 256, nullptr, nullptr, nullptr);
        gemm_wmma<EP_RESID_LN, A_F16_ROW><<<dim3(M / 32, 1, 1), 256, 0, stream>>>(
            nullptr, h_act, 2048, 1, 0, h_ffn2, ffn2_b,
            nxt, nullptr, 256, 1, 0, M, 256, 2048, f_mid, ln2_s, ln2_b);
        float* t = cur; cur = nxt; nxt = t;
    }

    // ---- outputs: (mask_feat, outs0..outs3) concatenated in d_out
    float* out = (float*)d_out;
    float* out_mask = out;                        // 2*256*4096
    float* out_lvl[4];
    out_lvl[0] = out + 2097152;
    out_lvl[1] = out_lvl[0] + 2097152;
    out_lvl[2] = out_lvl[1] + 524288;
    out_lvl[3] = out_lvl[2] + 131072;

    // mask_feat = 1x1 conv(outs0) + b; outs0 rows are the level-0 token rows
    gemm_wmma<EP_STORE, A_F32_ROW><<<dim3(4096 / 32, 1, 2), 256, 0, stream>>>(
        cur, nullptr, 256, 1, (long)S * 256, h_mask, mask_b,
        out_mask, nullptr, 1, 4096, 256L * 4096,
        4096, 256, 256, nullptr, nullptr, nullptr);
    for (int l = 0; l < 4; ++l)
        seq_to_chw_k<<<(2 * 256 * HW[l] + 255) / 256, 256, 0, stream>>>(
            cur, out_lvl[l], HW[l], starts[l]);
}
